// S4Blocks_50921132261667
// MI455X (gfx1250) — matrix operand order repaired and checked by reading, not compiled, not run on hardware
//
#include <hip/hip_runtime.h>
#include <hip/hip_bf16.h>
#include <math.h>

// ---------------- problem constants ----------------
constexpr int Bc   = 8;
constexpr int Lc   = 2048;
constexpr int Hc   = 512;            // d_model == GEMM N == GEMM K
constexpr int Nc   = 64;
constexpr int L2c  = 4096;           // 2*L FFT length
constexpr int NLc  = 4;              // total S4 layers
constexpr int Mrows = Bc * Lc;       // 16384 GEMM rows
constexpr float PI_F = 3.14159265358979323846f;

typedef __attribute__((ext_vector_type(16))) _Float16 v16h;
typedef __attribute__((ext_vector_type(8)))  _Float16 v8h;
typedef __attribute__((ext_vector_type(8)))  float    v8f;

union H2 { _Float16 h[2]; unsigned int u; };

// ---------------- complex helpers ----------------
__device__ __forceinline__ float2 cadd(float2 a, float2 b) { return make_float2(a.x + b.x, a.y + b.y); }
__device__ __forceinline__ float2 csub(float2 a, float2 b) { return make_float2(a.x - b.x, a.y - b.y); }
__device__ __forceinline__ float2 cmul(float2 a, float2 b) { return make_float2(a.x*b.x - a.y*b.y, a.x*b.y + a.y*b.x); }
__device__ __forceinline__ float2 cinv(float2 a) {
    float d = a.x*a.x + a.y*a.y;
    float r = 1.0f / d;
    return make_float2(a.x*r, -a.y*r);
}
__device__ __forceinline__ float2 cis(float ang) {
    float s, c;
    __sincosf(ang, &s, &c);
    return make_float2(c, s);
}

__device__ __forceinline__ float gelu_f(float x) {
    float x3 = x * x * x;
    return 0.5f * x * (1.0f + tanhf(0.7978845608028654f * (x + 0.044715f * x3)));
}
__device__ __forceinline__ float sigmoid_f(float x) { return 1.0f / (1.0f + expf(-x)); }

// ---------------- Stockham radix-2 DIF FFT (LDS-resident) ----------------
__device__ float2* stockham_fft(float2* x, float2* y, int n, float sign) {
    float2* src = x;
    float2* dst = y;
    int half = n >> 1;
    int s = 1;
    int m = half;
    while (s <= half) {
        for (int idx = threadIdx.x; idx < half; idx += blockDim.x) {
            int p = idx / s;
            int q = idx - p * s;
            float2 a = src[q + s * p];
            float2 b = src[q + s * (p + m)];
            float2 w = cis(sign * PI_F * (float)p / (float)m);
            dst[q + s * (2 * p)]     = cadd(a, b);
            dst[q + s * (2 * p + 1)] = cmul(csub(a, b), w);
        }
        __syncthreads();
        float2* t = src; src = dst; dst = t;
        s <<= 1; m >>= 1;
    }
    return src;
}

// ---------------- elementwise kernels (float4-vectorized) ----------------
__global__ void copy_kernel(const float4* __restrict__ src, float4* __restrict__ dst, int n4) {
    int i = blockIdx.x * blockDim.x + threadIdx.x;
    if (i < n4) dst[i] = src[i];
}

__global__ void glu_res_kernel(float4* __restrict__ x, const float4* __restrict__ g1,
                               const float4* __restrict__ g2, int n4) {
    int i = blockIdx.x * blockDim.x + threadIdx.x;
    if (i < n4) {
        float4 xv = x[i], a = g1[i], b = g2[i];
        xv.x += a.x * sigmoid_f(b.x);
        xv.y += a.y * sigmoid_f(b.y);
        xv.z += a.z * sigmoid_f(b.z);
        xv.w += a.w * sigmoid_f(b.w);
        x[i] = xv;
    }
}

__global__ void add_res_kernel(float4* __restrict__ x, const float4* __restrict__ g, int n4) {
    int i = blockIdx.x * blockDim.x + threadIdx.x;
    if (i < n4) {
        float4 xv = x[i], a = g[i];
        xv.x += a.x; xv.y += a.y; xv.z += a.z; xv.w += a.w;
        x[i] = xv;
    }
}

// ---------------- LayerNorm over H=512 (one row per block, 256 threads) ----------------
__global__ __launch_bounds__(256) void ln_kernel(const float* __restrict__ x,
                                                 const float* __restrict__ ls,
                                                 const float* __restrict__ lb,
                                                 float* __restrict__ out) {
    __shared__ float s1[256];
    __shared__ float s2[256];
    int row = blockIdx.x;
    const float* xr = x + (size_t)row * Hc;
    int t = threadIdx.x;
    float v0 = xr[t];
    float v1 = xr[t + 256];
    s1[t] = v0 + v1;
    s2[t] = v0 * v0 + v1 * v1;
    __syncthreads();
    for (int off = 128; off > 0; off >>= 1) {
        if (t < off) { s1[t] += s1[t + off]; s2[t] += s2[t + off]; }
        __syncthreads();
    }
    float mean = s1[0] * (1.0f / Hc);
    float var  = s2[0] * (1.0f / Hc) - mean * mean;
    float rstd = rsqrtf(var + 1e-6f);
    float* orow = out + (size_t)row * Hc;
    orow[t]       = (v0 - mean) * rstd * ls[t]       + lb[t];
    orow[t + 256] = (v1 - mean) * rstd * ls[t + 256] + lb[t + 256];
}

// ---------------- batched 32x32 tiled transpose: (B,R,C) -> (B,C,R) ----------------
__global__ __launch_bounds__(256) void transpose_kernel(const float* __restrict__ in,
                                                        float* __restrict__ out,
                                                        int R, int C) {
    __shared__ float tile[32][33];
    int b  = blockIdx.z;
    int c0 = blockIdx.x * 32;
    int r0 = blockIdx.y * 32;
    const float* inb = in + (size_t)b * R * C;
    float* outb = out + (size_t)b * R * C;
    for (int i = threadIdx.y; i < 32; i += 8)
        tile[i][threadIdx.x] = inb[(size_t)(r0 + i) * C + c0 + threadIdx.x];
    __syncthreads();
    for (int i = threadIdx.y; i < 32; i += 8)
        outb[(size_t)(c0 + i) * R + r0 + threadIdx.x] = tile[threadIdx.x][i];
}

// ---------------- Cauchy kernel eval + iFFT(L) + FFT(2L): one block per channel h ----------------
__global__ __launch_bounds__(256) void cauchy_fft_kernel(
        const float* __restrict__ Lre, const float* __restrict__ Lim,
        const float* __restrict__ Pre, const float* __restrict__ Pim,
        const float* __restrict__ Bre, const float* __restrict__ Bim,
        const float* __restrict__ Cre, const float* __restrict__ Cim,
        const float* __restrict__ logstep,
        float2* __restrict__ Kd) {
    __shared__ float2 bufA[L2c];
    __shared__ float2 bufB[L2c];

    int h = blockIdx.x;
    int t = threadIdx.x;

    // Stage per-n derived parameters into the (currently unused) pong buffer.
    float* par = reinterpret_cast<float*>(bufB);
    if (t < Nc) {
        int idx = h * Nc + t;
        float lr = fminf(Lre[idx], -1e-4f);
        float li = Lim[idx];
        float2 Pv  = make_float2(Pre[idx],  Pim[idx]);
        float2 Bv  = make_float2(Bre[idx],  Bim[idx]);
        float2 a0  = make_float2(Cre[idx], -Cim[idx]);   // conj(C)
        float2 a1  = make_float2(Pv.x,     -Pv.y);       // conj(P)
        float2 p00 = cmul(a0, Bv);
        float2 p01 = cmul(a0, Pv);
        float2 p10 = cmul(a1, Bv);
        float2 p11 = cmul(a1, Pv);
        float* o = par + t * 10;
        o[0] = lr;    o[1] = li;
        o[2] = p00.x; o[3] = p00.y;
        o[4] = p01.x; o[5] = p01.y;
        o[6] = p10.x; o[7] = p10.y;
        o[8] = p11.x; o[9] = p11.y;
    }
    __syncthreads();

    float step = expf(logstep[h]);
    float two_over_step = 2.0f / step;
    float2 one = make_float2(1.0f, 0.0f);

    for (int l = t; l < Lc; l += 256) {
        float2 Om   = cis(-2.0f * PI_F * (float)l / (float)Lc);
        float2 ipOm = cinv(cadd(one, Om));
        float2 g    = cmul(make_float2(two_over_step, 0.0f), cmul(csub(one, Om), ipOm));
        float2 c    = make_float2(2.0f * ipOm.x, 2.0f * ipOm.y);

        float2 k00 = make_float2(0.f, 0.f), k01 = k00, k10 = k00, k11 = k00;
        #pragma unroll 4
        for (int n = 0; n < Nc; ++n) {
            const float* o = par + n * 10;
            float2 lam = make_float2(o[0], o[1]);
            float2 r   = cinv(csub(g, lam));
            k00 = cadd(k00, cmul(make_float2(o[2], o[3]), r));
            k01 = cadd(k01, cmul(make_float2(o[4], o[5]), r));
            k10 = cadd(k10, cmul(make_float2(o[6], o[7]), r));
            k11 = cadd(k11, cmul(make_float2(o[8], o[9]), r));
        }
        float2 at = cmul(c, csub(k00, cmul(cmul(k01, cinv(cadd(one, k11))), k10)));
        bufA[l] = at;
    }
    __syncthreads();

    float2* R = stockham_fft(bufA, bufB, Lc, +1.0f);

    float2* W = (R == bufA) ? bufB : bufA;
    float scale = 1.0f / (float)Lc;
    for (int l = t; l < Lc; l += 256) W[l] = make_float2(R[l].x * scale, 0.0f);
    for (int l = Lc + t; l < L2c; l += 256) W[l] = make_float2(0.0f, 0.0f);
    __syncthreads();

    float2* F = stockham_fft(W, R, L2c, -1.0f);
    float2* out = Kd + (size_t)h * L2c;
    for (int f = t; f < L2c; f += 256) out[f] = F[f];
}

// ---------------- FFT conv per (b,h): y = irfft(rfft(h)*Kd) + D*h, then GELU ----------------
__global__ __launch_bounds__(256) void fft_conv_kernel(const float* __restrict__ hT,   // (B,H,L)
                                                       const float2* __restrict__ Kd,  // (H,4096)
                                                       const float* __restrict__ D,
                                                       float* __restrict__ yT) {       // (B,H,L)
    __shared__ float2 bufA[L2c];
    __shared__ float2 bufB[L2c];

    int bid = blockIdx.x;
    int b = bid / Hc;
    int h = bid - b * Hc;
    int t = threadIdx.x;

    const float* hin = hT + ((size_t)b * Hc + h) * Lc;
    for (int l = t; l < Lc; l += 256) bufA[l] = make_float2(hin[l], 0.0f);
    for (int l = Lc + t; l < L2c; l += 256) bufA[l] = make_float2(0.0f, 0.0f);
    __syncthreads();

    float2* F = stockham_fft(bufA, bufB, L2c, -1.0f);

    const float2* kd = Kd + (size_t)h * L2c;
    for (int f = t; f < L2c; f += 256) F[f] = cmul(F[f], kd[f]);
    __syncthreads();

    float2* G = (F == bufA) ? bufB : bufA;
    float2* Y = stockham_fft(F, G, L2c, +1.0f);

    float dh = D[h];
    float* yout = yT + ((size_t)b * Hc + h) * Lc;
    float scale = 1.0f / (float)L2c;
    for (int l = t; l < Lc; l += 256) {
        float v = Y[l].x * scale + dh * hin[l];
        yout[l] = gelu_f(v);
    }
}

// ---------------- WMMA GEMM: Out = act(A(Mx512) @ W(512x512) + bias) ----------------
// Block = 256 threads (8 waves), tile 128x64. Each wave owns one 16-col strip and
// four 16x16 sub-tiles (4 accumulators) sharing a single B fragment -> 4 WMMAs per
// K-step. LDS tiles padded to 80B row stride (16B-aligned b128 fragments, 20-bank
// lane stepping mod 64 -> conflict-free). Staging packs f16 pairs -> ds_store_b32.
constexpr int GK = Hc;         // GEMM K = 512
constexpr int GN = Hc;         // GEMM N = 512
constexpr int LDA = 40;        // padded LDS row stride in halves (80 bytes)

template <int ACT>
__global__ __launch_bounds__(256) void gemm_kernel(const float* __restrict__ A,
                                                   const float* __restrict__ W,
                                                   const float* __restrict__ bias,
                                                   float* __restrict__ Out) {
    __shared__ _Float16 As[128][LDA];   // [m][k]
    __shared__ _Float16 Bs[64][LDA];    // [n][k]  (transposed for contiguous lane K-runs)

    constexpr int nTilesN = GN >> 6;    // 8
    int bm = blockIdx.x / nTilesN;
    int bn = blockIdx.x - bm * nTilesN;
    int gm0 = bm << 7;                  // 128-row tile
    int gn0 = bn << 6;                  // 64-col tile

    int tid = threadIdx.x;
    int wave = tid >> 5;
    int lane = tid & 31;
    int laneHalf = lane >> 4;           // 0: lanes 0-15, 1: lanes 16-31
    int laneM = lane & 15;

    int sn = (wave & 3) << 4;           // wave's 16-col strip within tile
    int smBase = (wave >> 2) << 4;      // 0 or 16; rows smBase + {0,32,64,96}

    v8f acc[4] = {};

    for (int kt = 0; kt < GK; kt += 32) {
        // ---- stage A 128x32: each thread packs a K-pair (float2 load -> ds_store_b32)
        #pragma unroll
        for (int it = 0; it < 8; ++it) {
            int idx2 = tid + it * 256;            // 2048 K-pairs
            int r  = idx2 >> 4;                   // 0..127 (row)
            int c2 = idx2 & 15;                   // K-pair index -> k = 2*c2
            float2 av = *reinterpret_cast<const float2*>(
                            &A[(gm0 + r) * GK + kt + (c2 << 1)]);
            H2 p;
            p.h[0] = (_Float16)av.x;
            p.h[1] = (_Float16)av.y;
            *reinterpret_cast<unsigned int*>(&As[r][c2 << 1]) = p.u;
        }
        // ---- stage B 32x64 -> Bs[n][k]: pack K-pairs along k (2 coalesced b32 loads)
        #pragma unroll
        for (int it = 0; it < 4; ++it) {
            int idx2 = tid + it * 256;            // 1024 K-pairs
            int c  = idx2 & 63;                   // column n (coalesced across lanes)
            int rp = idx2 >> 6;                   // K-pair index -> k = 2*rp
            float w0 = W[(kt + (rp << 1))     * GN + gn0 + c];
            float w1 = W[(kt + (rp << 1) + 1) * GN + gn0 + c];
            H2 p;
            p.h[0] = (_Float16)w0;
            p.h[1] = (_Float16)w1;
            *reinterpret_cast<unsigned int*>(&Bs[c][rp << 1]) = p.u;
        }
        __syncthreads();

        // B fragment: col n = sn+laneM; elems 0-7 -> K kb..kb+7, elems 8-15 -> K 16+kb..
        const v8h* pb = reinterpret_cast<const v8h*>(&Bs[sn + laneM][laneHalf << 3]);
        v16h bf = __builtin_shufflevector(pb[0], pb[2],
                                          0, 1, 2, 3, 4, 5, 6, 7, 8, 9, 10, 11, 12, 13, 14, 15);

        // A fragments: row m = sm+laneM; elems 0-7 -> K kb..kb+7, elems 8-15 -> K 16+kb..
        int kb = laneHalf << 3;
        #pragma unroll
        for (int i = 0; i < 4; ++i) {
            int sm = smBase + (i << 5);
            const _Float16* rowp = &As[sm + laneM][0];
            v8h lo = *reinterpret_cast<const v8h*>(rowp + kb);
            v8h hi = *reinterpret_cast<const v8h*>(rowp + 16 + kb);
            v16h af = __builtin_shufflevector(lo, hi,
                                              0, 1, 2, 3, 4, 5, 6, 7, 8, 9, 10, 11, 12, 13, 14, 15);
            acc[i] = __builtin_amdgcn_wmma_f32_16x16x32_f16(false, af, false, bf,
                                                            (short)0, acc[i], false, false);
        }
        __syncthreads();
    }

    // Epilogue: C/D layout -> lane holds col laneM, rows laneHalf*8 + r of each sub-tile.
    int col = gn0 + sn + laneM;
    float bv = bias[col];
    #pragma unroll
    for (int i = 0; i < 4; ++i) {
        int row = gm0 + smBase + (i << 5) + (laneHalf << 3);
        float* outp = Out + row * GN + col;      // int32 index; r*GN folds to imm offsets
        #pragma unroll
        for (int r = 0; r < 8; ++r) {
            float v = acc[i][r] + bv;
            if (ACT == 1) v = gelu_f(v);
            outp[r * GN] = v;
        }
    }
}

// ---------------- host-side orchestration ----------------
extern "C" void kernel_launch(void* const* d_in, const int* in_sizes, int n_in,
                              void* d_out, int out_size, void* d_ws, size_t ws_size,
                              hipStream_t stream) {
    (void)in_sizes; (void)n_in; (void)out_size; (void)ws_size;

    const float* x_in  = (const float*)d_in[0];
    const float* sLre  = (const float*)d_in[1];
    const float* sLim  = (const float*)d_in[2];
    const float* sPre  = (const float*)d_in[3];
    const float* sPim  = (const float*)d_in[4];
    const float* sBre  = (const float*)d_in[5];
    const float* sBim  = (const float*)d_in[6];
    const float* sCre  = (const float*)d_in[7];
    const float* sCim  = (const float*)d_in[8];
    const float* sD    = (const float*)d_in[9];
    const float* sLog  = (const float*)d_in[10];
    const float* sb_ls = (const float*)d_in[11];
    const float* sb_lb = (const float*)d_in[12];
    const float* sb_W1 = (const float*)d_in[13];
    const float* sb_b1 = (const float*)d_in[14];
    const float* sb_W2 = (const float*)d_in[15];
    const float* sb_b2 = (const float*)d_in[16];
    const float* bk_ls = (const float*)d_in[17];
    const float* bk_lb = (const float*)d_in[18];
    const float* bk_W1 = (const float*)d_in[19];
    const float* bk_b1 = (const float*)d_in[20];
    const float* bk_W2 = (const float*)d_in[21];
    const float* bk_b2 = (const float*)d_in[22];

    const size_t SZ = (size_t)Bc * Lc * Hc;   // 8,388,608 floats per activation tensor
    float* ws   = (float*)d_ws;
    float* xbuf = ws;
    float* hN   = xbuf + SZ;
    float* hT   = hN + SZ;
    float* yT   = hT + SZ;
    float* ybuf = yT + SZ;
    float* g1   = ybuf + SZ;
    float* g2   = g1 + SZ;
    float2* Kd  = (float2*)(g2 + SZ);         // H * 4096 complex

    const int N4 = (int)(SZ / 4);
    const int EW_BLOCKS = (N4 + 255) / 256;
    const int GEMM_BLOCKS = (Mrows / 128) * (GN / 64);   // 1024

    copy_kernel<<<EW_BLOCKS, 256, 0, stream>>>((const float4*)x_in, (float4*)xbuf, N4);

    for (int i = 0; i < NLc; ++i) {
        const size_t pOff = (size_t)i * Hc * Nc;
        const size_t hOff = (size_t)i * Hc;
        const size_t wOff = (size_t)i * Hc * Hc;

        ln_kernel<<<Bc * Lc, 256, 0, stream>>>(xbuf, sb_ls + hOff, sb_lb + hOff, hN);
        transpose_kernel<<<dim3(Hc / 32, Lc / 32, Bc), dim3(32, 8), 0, stream>>>(hN, hT, Lc, Hc);
        cauchy_fft_kernel<<<Hc, 256, 0, stream>>>(
            sLre + pOff, sLim + pOff, sPre + pOff, sPim + pOff,
            sBre + pOff, sBim + pOff, sCre + pOff, sCim + pOff,
            sLog + hOff, Kd);
        fft_conv_kernel<<<Bc * Hc, 256, 0, stream>>>(hT, Kd, sD + hOff, yT);
        transpose_kernel<<<dim3(Lc / 32, Hc / 32, Bc), dim3(32, 8), 0, stream>>>(yT, ybuf, Hc, Lc);
        gemm_kernel<0><<<GEMM_BLOCKS, 256, 0, stream>>>(ybuf, sb_W1 + wOff, sb_b1 + hOff, g1);
        gemm_kernel<0><<<GEMM_BLOCKS, 256, 0, stream>>>(ybuf, sb_W2 + wOff, sb_b2 + hOff, g2);
        glu_res_kernel<<<EW_BLOCKS, 256, 0, stream>>>((float4*)xbuf, (const float4*)g1,
                                                      (const float4*)g2, N4);

        if ((i & 1) == 1) {
            int bb = i >> 1;
            const size_t bhOff = (size_t)bb * Hc;
            const size_t bwOff = (size_t)bb * Hc * Hc;
            ln_kernel<<<Bc * Lc, 256, 0, stream>>>(xbuf, bk_ls + bhOff, bk_lb + bhOff, hN);
            gemm_kernel<1><<<GEMM_BLOCKS, 256, 0, stream>>>(hN, bk_W1 + bwOff, bk_b1 + bhOff, g1);
            gemm_kernel<0><<<GEMM_BLOCKS, 256, 0, stream>>>(g1, bk_W2 + bwOff, bk_b2 + bhOff, g2);
            add_res_kernel<<<EW_BLOCKS, 256, 0, stream>>>((float4*)xbuf, (const float4*)g2, N4);
        }
    }

    copy_kernel<<<EW_BLOCKS, 256, 0, stream>>>((const float4*)xbuf, (float4*)d_out, N4);
}
